// SelfAttentionWithRelative_32899449487993
// MI455X (gfx1250) — compile-verified, matrix operand order checked by reading, not run on hardware
//
#include <hip/hip_runtime.h>
#include <hip/hip_bf16.h>

// ---------------- problem constants ----------------
constexpr int kB  = 64;        // batch
constexpr int kS  = 197;       // seq len
constexpr int kE  = 768;       // embed
constexpr int kH  = 12;        // heads
constexpr int kD  = 64;        // head dim
constexpr int kBH = kB * kH;   // 768
constexpr int kSP = 208;       // seq padded to mult of 16
constexpr int kM  = kB * kS;   // 12608 rows for projections

typedef _Float16 half_t;
typedef __attribute__((ext_vector_type(16))) _Float16     v16h;
typedef __attribute__((ext_vector_type(8)))  float        v8f;
typedef __attribute__((ext_vector_type(4)))  unsigned int u32x4;

struct HFrag { u32x4 a, b; };

__device__ __forceinline__ int imin(int a, int b) { return a < b ? a : b; }

// Load a 16x32 f16 WMMA A/B fragment from an LDS tile stored row-major
// [16][ldk].  Per ISA: lane<16 holds K {0..7,16..23} of row (lane&15);
// lane>=16 holds K {8..15,24..31}.  Two 16-byte ds_load_b128 per frag.
__device__ __forceinline__ v16h load_frag(const half_t* base, int ldk, int lane, int k0) {
  const half_t* p = base + (lane & 15) * ldk + k0 + ((lane >> 4) << 3);
  HFrag f;
  f.a = *(const u32x4*)p;
  f.b = *(const u32x4*)(p + 16);
  return __builtin_bit_cast(v16h, f);
}

// ---------------- shared 64x64 macro-tile GEMM core ----------------
// 128 threads = 4 waves; wave w owns output rows [16w,16w+16), all 64 cols.
// A element (r,k) at gA[r*sAr + k]; B^T element (n,k) at gB[n*sBr + k].
// K staged in 64-wide slabs, LDS row stride 72 halves (144B = 36 banks,
// gcd(36,64)=4 -> conflict-free).  Staging uses 16-byte b128 copies; all
// callers guarantee 16B alignment (bases/strides are multiples of 8 halves).
__device__ __forceinline__ void gemm_core(
    const half_t* gA, long sAr, int aRows,
    const half_t* gB, long sBr, int bRows,
    int kTot, int kMax, half_t* lA, half_t* lB, v8f* acc) {
  const int tid  = threadIdx.x;
  const int wave = tid >> 5, lane = tid & 31;
  const u32x4 z4 = {0, 0, 0, 0};
  for (int k0 = 0; k0 < kTot; k0 += 64) {
    if (k0 + 64 < kTot) {            // hint next slab -> global_prefetch_b8
      __builtin_prefetch(gA + (long)(k0 + 64), 0, 1);
      __builtin_prefetch(gB + (long)(k0 + 64), 0, 1);
    }
    // stage 64x64 A and B^T slabs, 16B per transfer, 4 iters/thread
    for (int t = tid; t < 512; t += 128) {
      int r = t >> 3, c8 = (t & 7) << 3, k = k0 + c8;
      half_t* dA = lA + r * 72 + c8;
      half_t* dB = lB + r * 72 + c8;
      if (k + 8 <= kMax) {
        if (r < aRows) *(u32x4*)dA = *(const u32x4*)(gA + (long)r * sAr + k);
        else           *(u32x4*)dA = z4;
        if (r < bRows) *(u32x4*)dB = *(const u32x4*)(gB + (long)r * sBr + k);
        else           *(u32x4*)dB = z4;
      } else {
        for (int j = 0; j < 8; ++j) {
          int kk = k + j;
          dA[j] = (r < aRows && kk < kMax) ? gA[(long)r * sAr + kk] : (half_t)0.f;
          dB[j] = (r < bRows && kk < kMax) ? gB[(long)r * sBr + kk] : (half_t)0.f;
        }
      }
    }
    __syncthreads();
    const half_t* aBase = lA + (wave << 4) * 72;
#pragma unroll
    for (int kk = 0; kk < 64; kk += 32) {
      v16h af = load_frag(aBase, 72, lane, kk);
#pragma unroll
      for (int nt = 0; nt < 4; ++nt) {
        v16h bf = load_frag(lB + (nt << 4) * 72, 72, lane, kk);
        acc[nt] = __builtin_amdgcn_wmma_f32_16x16x32_f16(
            false, af, false, bf, (short)0, acc[nt], false, false);
      }
    }
    __syncthreads();
  }
}

#define GEMM_SHARED                                    \
  __shared__ alignas(16) half_t lA[64 * 72];           \
  __shared__ alignas(16) half_t lB[64 * 72];           \
  v8f acc[4];                                          \
  { v8f z = {}; for (int i = 0; i < 4; ++i) acc[i] = z; }

// C/D VGPR layout: vgpr i, lane l -> row = i + 8*(l>>4), col = l&15.
#define EPILOGUE_VARS                                  \
  const int lane = threadIdx.x & 31;                   \
  const int rbase = ((threadIdx.x >> 5) << 4) + ((lane >> 4) << 3); \
  const int cbase = lane & 15;

// ---------------- elementwise kernels ----------------
__global__ void k_f2h(const float* __restrict__ in, half_t* __restrict__ out, int n) {
  int i = blockIdx.x * blockDim.x + threadIdx.x;
  if (i < n) out[i] = (half_t)in[i];
}

// Build q_pos (S,S,64) and v_pos^T (S,64,208) f16 via table gathers.
__global__ void k_pos(const float* qtx, const float* qty,
                      const float* vtx, const float* vty,
                      const int* xd, const int* yd,
                      half_t* qpos, half_t* vposT) {
  int i = blockIdx.x * blockDim.x + threadIdx.x;
  if (i >= kS * kS) return;
  int dx = xd[i], dy = yd[i];
  int q = i / kS, kk = i % kS;
  size_t qb = (size_t)i * 64;
  size_t vb = (size_t)q * kD * kSP + kk;
  for (int j = 0; j < 32; ++j) {
    qpos[qb + j]      = (half_t)qtx[dx * 32 + j];
    qpos[qb + 32 + j] = (half_t)qty[dy * 32 + j];
    vposT[vb + (size_t)j * kSP]        = (half_t)vtx[dx * 32 + j];
    vposT[vb + (size_t)(32 + j) * kSP] = (half_t)vty[dy * 32 + j];
  }
}

// ---------------- phase kernels ----------------
// QKV projection: (12608x768) @ W^T + b.
// Q,K stored [bh][s][d]; V stored transposed [bh][d][sP] so the P@V GEMM
// has unit-stride K and stages with b128 copies.
__global__ void k_proj(const half_t* __restrict__ xh,
                       const half_t* wq, const half_t* wk, const half_t* wv,
                       const float* bq, const float* bk, const float* bv,
                       half_t* Q, half_t* K, half_t* Vt) {
  GEMM_SHARED
  int m0 = blockIdx.x << 6, n0 = blockIdx.y << 6, sel = blockIdx.z;
  const half_t* W    = sel == 0 ? wq : sel == 1 ? wk : wv;
  const float*  bias = sel == 0 ? bq : sel == 1 ? bk : bv;
  gemm_core(xh + (long)m0 * kE, kE, imin(64, kM - m0),
            W + (long)n0 * kE, kE, imin(64, kE - n0),
            kE, kE, lA, lB, acc);
  EPILOGUE_VARS
#pragma unroll
  for (int nt = 0; nt < 4; ++nt)
#pragma unroll
    for (int i = 0; i < 8; ++i) {
      int m = m0 + rbase + i, n = n0 + (nt << 4) + cbase;
      if (m < kM) {
        int b = m / kS, s = m % kS, h = n >> 6, d = n & 63;
        half_t val = (half_t)(acc[nt][i] + bias[n]);
        size_t bh = (size_t)b * kH + h;
        if (sel == 0)      Q[(bh * kS + s) * kD + d] = val;
        else if (sel == 1) K[(bh * kS + s) * kD + d] = val;
        else               Vt[(bh * kD + d) * kSP + s] = val;
      }
    }
}

// Rel logits (per q): Rel_q[bh][k] = Q[bh,q,:] . q_pos[q,k,:]  -> P buffer [q][bh][k]
__global__ void k_rel(const half_t* __restrict__ Q,
                      const half_t* __restrict__ qpos,
                      half_t* __restrict__ P) {
  GEMM_SHARED
  int m0 = blockIdx.x << 6, n0 = blockIdx.y << 6, q = blockIdx.z;
  gemm_core(Q + (size_t)m0 * kS * kD + (size_t)q * kD, (long)kS * kD, 64,
            qpos + ((size_t)q * kS + n0) * kD, kD, imin(64, kS - n0),
            kD, kD, lA, lB, acc);
  EPILOGUE_VARS
#pragma unroll
  for (int nt = 0; nt < 4; ++nt)
#pragma unroll
    for (int i = 0; i < 8; ++i) {
      int m = m0 + rbase + i, n = n0 + (nt << 4) + cbase;
      if (n < kSP)
        P[((size_t)q * kBH + m) * kSP + n] = (half_t)acc[nt][i];
    }
}

// Content logits + rel add + softmax, in-place over P.  One block = (bh, 16 q rows, all k).
__global__ void k_attn(const half_t* __restrict__ Q, const half_t* __restrict__ K,
                       half_t* __restrict__ P) {
  __shared__ alignas(16) half_t lQ[16 * 72];
  __shared__ alignas(16) half_t lK[kSP * 72];
  __shared__ float lgt[16 * 224];
  const int q0 = blockIdx.x << 4, bh = blockIdx.y;
  const int tid = threadIdx.x, wave = tid >> 5, lane = tid & 31;
  const u32x4 z4 = {0, 0, 0, 0};

  for (int t = tid; t < 16 * 8; t += 128) {
    int r = t >> 3, c8 = (t & 7) << 3, q = q0 + r;
    u32x4 v = z4;
    if (q < kS) v = *(const u32x4*)(Q + ((size_t)bh * kS + q) * kD + c8);
    *(u32x4*)(lQ + r * 72 + c8) = v;
  }
  for (int t = tid; t < kSP * 8; t += 128) {
    int r = t >> 3, c8 = (t & 7) << 3;
    u32x4 v = z4;
    if (r < kS) v = *(const u32x4*)(K + ((size_t)bh * kS + r) * kD + c8);
    *(u32x4*)(lK + r * 72 + c8) = v;
  }
  __syncthreads();

  v16h af0 = load_frag(lQ, 72, lane, 0);
  v16h af1 = load_frag(lQ, 72, lane, 32);
  const int rb = (lane >> 4) << 3, cb = lane & 15;
  for (int nt = wave; nt < 13; nt += 4) {
    v8f a = {};
    v16h bf0 = load_frag(lK + nt * 16 * 72, 72, lane, 0);
    v16h bf1 = load_frag(lK + nt * 16 * 72, 72, lane, 32);
    a = __builtin_amdgcn_wmma_f32_16x16x32_f16(false, af0, false, bf0, (short)0, a, false, false);
    a = __builtin_amdgcn_wmma_f32_16x16x32_f16(false, af1, false, bf1, (short)0, a, false, false);
    int col = (nt << 4) + cb;
#pragma unroll
    for (int i = 0; i < 8; ++i) {
      int row = rb + i, q = q0 + row;
      float relv = 0.f;
      if (q < kS && col < kSP) relv = (float)P[((size_t)q * kBH + bh) * kSP + col];
      lgt[row * 224 + col] = (a[i] + relv) * 0.125f;   // 1/sqrt(64)
    }
  }
  __syncthreads();

  // softmax: 8 threads per row
  int r = tid >> 3, t8 = tid & 7;
  float mx = -3.0e38f;
  for (int k = t8; k < kS; k += 8) mx = fmaxf(mx, lgt[r * 224 + k]);
  for (int off = 4; off; off >>= 1) mx = fmaxf(mx, __shfl_xor(mx, off, 8));
  float sum = 0.f;
  for (int k = t8; k < kS; k += 8) {
    float e = __expf(lgt[r * 224 + k] - mx);
    lgt[r * 224 + k] = e;
    sum += e;
  }
  for (int off = 4; off; off >>= 1) sum += __shfl_xor(sum, off, 8);
  float inv = 1.f / sum;
  int q = q0 + r;
  if (q < kS)
    for (int k = t8; k < kSP; k += 8) {
      float p = (k < kS) ? lgt[r * 224 + k] * inv : 0.f;
      P[((size_t)q * kBH + bh) * kSP + k] = (half_t)p;
    }
}

// Content output (per bh): O1[b,q,h*64+d] = P[bh] @ V[bh]   (fp32)
__global__ void k_pv(const half_t* __restrict__ P, const half_t* __restrict__ Vt,
                     float* __restrict__ O1) {
  GEMM_SHARED
  int m0 = blockIdx.x << 6, bh = blockIdx.y;
  gemm_core(P + ((size_t)m0 * kBH + bh) * kSP, (long)kBH * kSP, imin(64, kS - m0),
            Vt + (size_t)bh * kD * kSP, kSP, 64,
            224, kS, lA, lB, acc);
  EPILOGUE_VARS
  int b = bh / kH, h = bh % kH;
#pragma unroll
  for (int nt = 0; nt < 4; ++nt)
#pragma unroll
    for (int i = 0; i < 8; ++i) {
      int q = m0 + rbase + i, d = (nt << 4) + cbase;
      if (q < kS)
        O1[((size_t)b * kS + q) * kE + h * kD + d] = acc[nt][i];
    }
}

// v_pos output (per q): add P_qslice @ v_pos[q]; finalize O1 -> f16 for out-proj.
__global__ void k_vpos(const half_t* __restrict__ P, const half_t* __restrict__ vposT,
                       const float* __restrict__ O1, half_t* __restrict__ O1h) {
  GEMM_SHARED
  int m0 = blockIdx.x << 6, q = blockIdx.y;
  gemm_core(P + ((size_t)q * kBH + m0) * kSP, kSP, 64,
            vposT + (size_t)q * kD * kSP, kSP, 64,
            224, kS, lA, lB, acc);
  EPILOGUE_VARS
#pragma unroll
  for (int nt = 0; nt < 4; ++nt)
#pragma unroll
    for (int i = 0; i < 8; ++i) {
      int bh = m0 + rbase + i, d = (nt << 4) + cbase;
      int b = bh / kH, h = bh % kH;
      size_t idx = ((size_t)b * kS + q) * kE + h * kD + d;
      O1h[idx] = (half_t)(acc[nt][i] + O1[idx]);
    }
}

// Output projection: out = O1h @ Wo^T + bo   (fp32 out)
__global__ void k_out(const half_t* __restrict__ O1h, const half_t* __restrict__ wo,
                      const float* __restrict__ bo, float* __restrict__ out) {
  GEMM_SHARED
  int m0 = blockIdx.x << 6, n0 = blockIdx.y << 6;
  gemm_core(O1h + (long)m0 * kE, kE, imin(64, kM - m0),
            wo + (long)n0 * kE, kE, imin(64, kE - n0),
            kE, kE, lA, lB, acc);
  EPILOGUE_VARS
#pragma unroll
  for (int nt = 0; nt < 4; ++nt)
#pragma unroll
    for (int i = 0; i < 8; ++i) {
      int m = m0 + rbase + i, n = n0 + (nt << 4) + cbase;
      if (m < kM)
        out[(size_t)m * kE + n] = acc[nt][i] + bo[n];
    }
}

// ---------------- host launcher ----------------
extern "C" void kernel_launch(void* const* d_in, const int* in_sizes, int n_in,
                              void* d_out, int out_size, void* d_ws, size_t ws_size,
                              hipStream_t stream) {
  (void)in_sizes; (void)n_in; (void)out_size; (void)ws_size;
  const float* x   = (const float*)d_in[0];
  const float* Wq  = (const float*)d_in[1];
  const float* bq  = (const float*)d_in[2];
  const float* Wk  = (const float*)d_in[3];
  const float* bk  = (const float*)d_in[4];
  const float* Wv  = (const float*)d_in[5];
  const float* bv  = (const float*)d_in[6];
  const float* Wo  = (const float*)d_in[7];
  const float* bo  = (const float*)d_in[8];
  const float* qtx = (const float*)d_in[9];
  const float* qty = (const float*)d_in[10];
  const float* vtx = (const float*)d_in[11];
  const float* vty = (const float*)d_in[12];
  const int*   xd  = (const int*)d_in[13];
  const int*   yd  = (const int*)d_in[14];
  float* out = (float*)d_out;

  char* ws = (char*)d_ws;
  size_t off = 0;
  auto take = [&](size_t bytes) {
    void* p = ws + off;
    off += (bytes + 255) & ~(size_t)255;
    return p;
  };
  const size_t NX   = (size_t)kM * kE;           // 9.68M
  const size_t NW   = (size_t)kE * kE;
  const size_t NQ   = (size_t)kBH * kS * kD;     // 9.68M
  const size_t NVT  = (size_t)kBH * kD * kSP;    // 10.2M
  const size_t NPOS = (size_t)kS * kS * kD;
  const size_t NVPT = (size_t)kS * kD * kSP;     // 2.62M
  const size_t NP   = (size_t)kS * kBH * kSP;    // 31.5M

  half_t* xh    = (half_t*)take(NX * 2);
  half_t* wqh   = (half_t*)take(NW * 2);
  half_t* wkh   = (half_t*)take(NW * 2);
  half_t* wvh   = (half_t*)take(NW * 2);
  half_t* woh   = (half_t*)take(NW * 2);
  half_t* Qh    = (half_t*)take(NQ * 2);
  half_t* Kh    = (half_t*)take(NQ * 2);
  half_t* Vt    = (half_t*)take(NVT * 2);
  half_t* qpos  = (half_t*)take(NPOS * 2);
  half_t* vposT = (half_t*)take(NVPT * 2);
  half_t* P     = (half_t*)take(NP * 2);
  float*  O1    = (float*)take(NX * 4);
  half_t* O1h   = xh;  // xh dead after k_proj; reuse

  k_f2h<<<(int)((NX + 255) / 256), 256, 0, stream>>>(x, xh, (int)NX);
  k_f2h<<<(int)((NW + 255) / 256), 256, 0, stream>>>(Wq, wqh, (int)NW);
  k_f2h<<<(int)((NW + 255) / 256), 256, 0, stream>>>(Wk, wkh, (int)NW);
  k_f2h<<<(int)((NW + 255) / 256), 256, 0, stream>>>(Wv, wvh, (int)NW);
  k_f2h<<<(int)((NW + 255) / 256), 256, 0, stream>>>(Wo, woh, (int)NW);
  k_pos<<<(kS * kS + 255) / 256, 256, 0, stream>>>(qtx, qty, vtx, vty, xd, yd, qpos, vposT);

  k_proj<<<dim3(kM / 64, kE / 64, 3), 128, 0, stream>>>(
      xh, wqh, wkh, wvh, bq, bk, bv, Qh, Kh, Vt);
  k_rel<<<dim3(kBH / 64, 4, kS), 128, 0, stream>>>(Qh, qpos, P);
  k_attn<<<dim3((kS + 15) / 16, kBH), 128, 0, stream>>>(Qh, Kh, P);
  k_pv<<<dim3((kS + 63) / 64, kBH), 128, 0, stream>>>(P, Vt, O1);
  k_vpos<<<dim3(kBH / 64, kS), 128, 0, stream>>>(P, vposT, O1, O1h);
  k_out<<<dim3(kM / 64, kE / 64), 128, 0, stream>>>(O1h, woh, bo, out);
}
